// NMSWTokenSegFormer3D_82051055223232
// MI455X (gfx1250) — matrix operand order repaired and checked by reading, not compiled, not run on hardware
//
#include <hip/hip_runtime.h>
#include <cstdint>

// ---------------------------------------------------------------------------
// MI455X / gfx1250: bf16 WMMA GEMMs (v_wmma_f32_16x16x32_bf16, wave32),
// async global->LDS staging (ASYNCcnt), fused bias/GELU/residual epilogues.
// ---------------------------------------------------------------------------

#define USE_ASYNC 1

typedef __bf16 bf16_t;
typedef __attribute__((ext_vector_type(16))) __bf16 v16bf;
typedef __attribute__((ext_vector_type(8)))  __bf16 bf16x8;
typedef __attribute__((ext_vector_type(4)))  __bf16 bf16x4;
typedef __attribute__((ext_vector_type(8)))  float  v8f;

__device__ __forceinline__ float gelu_f(float v) {
  return 0.5f * v * (1.0f + erff(v * 0.7071067811865476f));
}

// epilogue flags
#define FL_ACC  1   // C[idx] += result (residual fuse, f32)
#define FL_GELU 2   // apply exact GELU
#define FL_BF16 4   // store bf16 result to Cb
#define FL_F32  8   // store f32 result to C

// ---------------- WMMA GEMM -------------------------------------------------
// C[M,N] = A[M,K](bf16) * B(bf16) + bias[n], epilogue per flags.
// Requirements (guaranteed by all call sites): M%64==0, N%32==0, K%32==0.
//   A(m,k) = A[m*lda + k]                        (row-major, 16B-aligned rows)
//   bMode0: B(k,n) = B[k*ldb + n]                (row-major K x N, pre-packed)
//   bMode1: B(k,n) = B[n*ldbN + k]               (k-contiguous, e.g. K^T)
// Batched over grid.z: z = zb*heads + zh, per-operand (zb,zh) element strides.
#define GTM 64
#define GTN 32
#define GTK 32

__global__ __launch_bounds__(256) void k_wmma_gemm(
    const bf16_t* __restrict__ A, const bf16_t* __restrict__ Bm,
    float* __restrict__ C, bf16_t* __restrict__ Cb,
    const float* __restrict__ bias,
    int M, int N, int K, int lda, int ldb, int ldbN, int ldc,
    long aZb, long aZh, long bZb, long bZh, long cZb, long cZh,
    int heads, int flags, int bMode)
{
  __shared__ __align__(16) bf16_t As[GTM][GTK];   // 4 KB
  __shared__ __align__(16) bf16_t Bs[GTK][GTN];   // 2 KB
  int z  = blockIdx.z;
  int zb = z / heads, zh = z - zb * heads;
  A  += aZb * zb + aZh * zh;
  Bm += bZb * zb + bZh * zh;
  if (C)  C  += cZb * zb + cZh * zh;
  if (Cb) Cb += cZb * zb + cZh * zh;

  int m0 = blockIdx.y * GTM;
  int n0 = blockIdx.x * GTN;
  int t    = threadIdx.x;
  int lane = t & 31;
  int wave = t >> 5;         // 8 waves: 4 (M) x 2 (N)
  int wm = wave >> 1;
  int wn = wave & 1;
  int hl = lane >> 4;        // lane half per ISA 16x16x32 layout

  v8f acc = {0.f, 0.f, 0.f, 0.f, 0.f, 0.f, 0.f, 0.f};

  for (int kb = 0; kb < K; kb += GTK) {
    // ---- stage A tile 64x32: one b128 per thread, async to LDS ----
    {
      int r  = t >> 2;
      int k0 = (t & 3) * 8;
      const bf16_t* ap = A + (long)(m0 + r) * lda + kb + k0;
#if USE_ASYNC
      unsigned lofs = (unsigned)(uintptr_t)&As[r][k0];
      asm volatile("global_load_async_to_lds_b128 %0, %1, off"
                   :: "v"(lofs), "v"(ap) : "memory");
#else
      *(bf16x8*)&As[r][k0] = *(const bf16x8*)ap;
#endif
    }
    // ---- stage B tile 32x32 ----
    if (bMode == 0) {
      int kk = t >> 3;             // 0..31
      int nn = (t & 7) * 4;        // 0,4,..,28
      const bf16_t* bp = Bm + (long)(kb + kk) * ldb + n0 + nn;
      if (kb + GTK < K) __builtin_prefetch(bp + (long)GTK * ldb, 0, 1);
      *(bf16x4*)&Bs[kk][nn] = *(const bf16x4*)bp;
    } else {
      int nn = t >> 3;             // 0..31 (column of tile)
      int kc = (t & 7) * 4;        // 0,4,..,28 (k chunk, contiguous in mem)
      const bf16_t* bp = Bm + (long)(n0 + nn) * ldbN + kb + kc;
      bf16x4 v4 = *(const bf16x4*)bp;
#pragma unroll
      for (int j = 0; j < 4; ++j) Bs[kc + j][nn] = v4[j];
    }
#if USE_ASYNC
    asm volatile("s_wait_asynccnt 0x0" ::: "memory");
#endif
    __syncthreads();
    // fragments per ISA 7.12.2: A row-per-lane with 8*half K swizzle,
    // B column-per-lane (n = lane&15) with the same 8*half K swizzle.
    union { v16bf v; bf16x8 h[2]; } fa, fb;
    int mf = wm * 16 + (lane & 15);
    fa.h[0] = *(const bf16x8*)&As[mf][8 * hl];
    fa.h[1] = *(const bf16x8*)&As[mf][16 + 8 * hl];
#pragma unroll
    for (int s = 0; s < 16; ++s)
      fb.v[s] = Bs[(s >> 3) * 16 + 8 * hl + (s & 7)][wn * 16 + (lane & 15)];
    acc = __builtin_amdgcn_wmma_f32_16x16x32_bf16(false, fa.v, false, fb.v,
                                                  (short)0, acc, false, false);
    __syncthreads();
  }

  // ---- epilogue: C/D layout VGPR v -> row (v + 8*half), col lane&15 ----
  int nG = n0 + wn * 16 + (lane & 15);
  float bv = bias ? bias[nG] : 0.0f;
#pragma unroll
  for (int v = 0; v < 8; ++v) {
    int  mG  = m0 + wm * 16 + v + 8 * hl;
    long idx = (long)mG * ldc + nG;
    float r = acc[v] + bv;
    if (flags & FL_ACC)  r += C[idx];
    if (flags & FL_GELU) r = gelu_f(r);
    if (flags & FL_F32)  C[idx] = r;
    if (flags & FL_BF16) Cb[idx] = (bf16_t)r;
  }
}

// ---------------- packing kernels -------------------------------------------

// W[N,K] f32 -> Wt[Kpad,N] bf16 (transpose + zero pad); batched via blockIdx.y
__global__ void k_wt(const float* __restrict__ w, bf16_t* __restrict__ wt,
                     int N, int K, int Kpad)
{
  long tot = (long)Kpad * N;
  long i = (long)blockIdx.x * 256 + threadIdx.x;
  if (i >= tot) return;
  const float* wz = w + (long)blockIdx.y * N * K;
  bf16_t* wtz = wt + (long)blockIdx.y * tot;
  int n = (int)(i % N);
  int k = (int)(i / N);
  wtz[i] = (k < K) ? (bf16_t)wz[(long)n * K + k] : (bf16_t)0.0f;
}

// coords [BK,3] f32 -> [BK,32] bf16 zero-padded
__global__ void k_pad_coords(const float* __restrict__ c, bf16_t* __restrict__ y, int n)
{
  long i = (long)blockIdx.x * 256 + threadIdx.x;
  if (i >= (long)n * 32) return;
  int k = (int)(i & 31);
  long r = i >> 5;
  y[i] = (k < 3) ? (bf16_t)c[r * 3 + k] : (bf16_t)0.0f;
}

// ---------------- elementwise / reduction kernels ---------------------------

__global__ void k_add(const float* __restrict__ a, const float* __restrict__ b,
                      float* __restrict__ y, long n) {
  long i = (long)blockIdx.x * 256 + threadIdx.x;
  if (i < n) y[i] = a[i] + b[i];
}

// LayerNorm over E=256 (one block per row); optional f32 / bf16 outputs
__global__ __launch_bounds__(256) void k_layernorm(
    const float* __restrict__ x, const float* __restrict__ g,
    const float* __restrict__ b, float* __restrict__ outF,
    bf16_t* __restrict__ outB)
{
  __shared__ float red[256];
  long row = blockIdx.x;
  int  t   = threadIdx.x;
  float v = x[row * 256 + t];
  red[t] = v; __syncthreads();
  for (int st = 128; st > 0; st >>= 1) {
    if (t < st) red[t] += red[t + st];
    __syncthreads();
  }
  float m = red[0] * (1.0f / 256.0f); __syncthreads();
  float d = v - m;
  red[t] = d * d; __syncthreads();
  for (int st = 128; st > 0; st >>= 1) {
    if (t < st) red[t] += red[t + st];
    __syncthreads();
  }
  float var = red[0] * (1.0f / 256.0f);
  float y = d * rsqrtf(var + 1e-5f) * g[t] + b[t];
  if (outF) outF[row * 256 + t] = y;
  if (outB) outB[row * 256 + t] = (bf16_t)y;
}

// softmax over rows of length 1024 (4 elems/thread), scaled; bf16 out
__global__ __launch_bounds__(256) void k_softmax(
    const float* __restrict__ s, bf16_t* __restrict__ p, float scale)
{
  __shared__ float red[256];
  long row = blockIdx.x;
  const float* r = s + row * 1024L;
  bf16_t* o = p + row * 1024L;
  int t = threadIdx.x;
  float v[4];
  float mx = -3.4e38f;
#pragma unroll
  for (int j = 0; j < 4; ++j) { v[j] = r[t + j * 256] * scale; mx = fmaxf(mx, v[j]); }
  red[t] = mx; __syncthreads();
  for (int st = 128; st > 0; st >>= 1) {
    if (t < st) red[t] = fmaxf(red[t], red[t + st]);
    __syncthreads();
  }
  mx = red[0]; __syncthreads();
  float sum = 0.f;
#pragma unroll
  for (int j = 0; j < 4; ++j) { v[j] = __expf(v[j] - mx); sum += v[j]; }
  red[t] = sum; __syncthreads();
  for (int st = 128; st > 0; st >>= 1) {
    if (t < st) red[t] += red[t + st];
    __syncthreads();
  }
  float inv = 1.0f / red[0];
#pragma unroll
  for (int j = 0; j < 4; ++j) o[t + j * 256] = (bf16_t)(v[j] * inv);
}

// BatchNorm training-mode stats: x layout [N, C, S]; one block per channel
__global__ __launch_bounds__(256) void k_bn_stats(
    const float* __restrict__ x, float* __restrict__ mv, int C, long S, long Nn)
{
  __shared__ float rs[256], rs2[256];
  int c = blockIdx.x, t = threadIdx.x;
  long tot = Nn * S;
  float s = 0.f, s2 = 0.f;
  for (long i = t; i < tot; i += 256) {
    long n = i / S, sp = i - n * S;
    float v = x[(n * (long)C + c) * S + sp];
    s += v; s2 += v * v;
  }
  rs[t] = s; rs2[t] = s2; __syncthreads();
  for (int st = 128; st > 0; st >>= 1) {
    if (t < st) { rs[t] += rs[t + st]; rs2[t] += rs2[t + st]; }
    __syncthreads();
  }
  if (t == 0) {
    float m = rs[0] / (float)tot;
    mv[c] = m;
    mv[C + c] = rs2[0] / (float)tot - m * m;
  }
}

// BN normalize + GELU in place; optional bf16 copy out
__global__ void k_bn_gelu(float* __restrict__ x, const float* __restrict__ mv,
                          const float* __restrict__ g, const float* __restrict__ b,
                          bf16_t* __restrict__ yb, int C, long S, long total)
{
  long i = (long)blockIdx.x * 256 + threadIdx.x;
  if (i >= total) return;
  int c = (int)((i / S) % C);
  float m = mv[c], v = mv[C + c];
  float t = (x[i] - m) * rsqrtf(v + 1e-5f) * g[c] + b[c];
  float r = gelu_f(t);
  x[i] = r;
  if (yb) yb[i] = (bf16_t)r;
}

// conv3d kernel=3 stride=2 pad=1 (NCDHW, OIDHW weights)
__global__ void k_conv3(const float* __restrict__ x, const float* __restrict__ w,
                        const float* __restrict__ bia, float* __restrict__ y,
                        int N, int Cin, int Din, int Cout)
{
  int Dout = Din >> 1;
  long total = (long)N * Cout * Dout * Dout * Dout;
  long i = (long)blockIdx.x * 256 + threadIdx.x;
  if (i >= total) return;
  int ow = (int)(i % Dout); long r = i / Dout;
  int oh = (int)(r % Dout); r /= Dout;
  int od = (int)(r % Dout); r /= Dout;
  int oc = (int)(r % Cout); int n = (int)(r / Cout);
  float s = bia[oc];
  for (int ci = 0; ci < Cin; ++ci) {
    const float* xp = x + ((long)n * Cin + ci) * Din * Din * Din;
    const float* wp = w + ((long)oc * Cin + ci) * 27;
    for (int kd = 0; kd < 3; ++kd) {
      int id = od * 2 - 1 + kd; if ((unsigned)id >= (unsigned)Din) continue;
      for (int kh = 0; kh < 3; ++kh) {
        int ih = oh * 2 - 1 + kh; if ((unsigned)ih >= (unsigned)Din) continue;
        for (int kw = 0; kw < 3; ++kw) {
          int iw = ow * 2 - 1 + kw; if ((unsigned)iw >= (unsigned)Din) continue;
          s += xp[(id * Din + ih) * Din + iw] * wp[(kd * 3 + kh) * 3 + kw];
        }
      }
    }
  }
  y[i] = s;
}

// conv3d kernel=2 stride=2 pad=0 on 2^3 input -> 1^3 output
__global__ void k_conv_k2(const float* __restrict__ x, const float* __restrict__ w,
                          const float* __restrict__ bia, float* __restrict__ y,
                          int N, int Cin, int Cout)
{
  long total = (long)N * Cout;
  long i = (long)blockIdx.x * 256 + threadIdx.x;
  if (i >= total) return;
  int oc = (int)(i % Cout); int n = (int)(i / Cout);
  float s = bia[oc];
  for (int ci = 0; ci < Cin; ++ci) {
    const float* xp = x + ((long)n * Cin + ci) * 8;
    const float* wp = w + ((long)oc * Cin + ci) * 8;
#pragma unroll
    for (int j = 0; j < 8; ++j) s += xp[j] * wp[j];
  }
  y[i] = s;
}

// ConvTranspose3d kernel=2 stride=2 pad=0; torch weight layout [Cin,Cout,2,2,2]
__global__ void k_deconv2x(const float* __restrict__ x, const float* __restrict__ w,
                           const float* __restrict__ bia, float* __restrict__ y,
                           int N, int Cin, int Din, int Cout)
{
  int Dout = Din * 2;
  long total = (long)N * Cout * Dout * Dout * Dout;
  long i = (long)blockIdx.x * 256 + threadIdx.x;
  if (i >= total) return;
  int ox = (int)(i % Dout); long r = i / Dout;
  int oy = (int)(r % Dout); r /= Dout;
  int oz = (int)(r % Dout); r /= Dout;
  int oc = (int)(r % Cout); int n = (int)(r / Cout);
  int ix = ox >> 1, ax = ox & 1;
  int iy = oy >> 1, ay = oy & 1;
  int iz = oz >> 1, az = oz & 1;
  int widx = az * 4 + ay * 2 + ax;
  float s = bia[oc];
  for (int ci = 0; ci < Cin; ++ci) {
    s += x[((long)n * Cin + ci) * Din * Din * Din + (iz * Din + iy) * Din + ix]
       * w[((long)ci * Cout + oc) * 8 + widx];
  }
  y[i] = s;
}

// final 1x1x1 conv: [N,32,512] -> [N,2,512]
__global__ void k_outconv(const float* __restrict__ f, const float* __restrict__ w,
                          const float* __restrict__ bia, float* __restrict__ y, int N)
{
  long total = (long)N * 2 * 512;
  long i = (long)blockIdx.x * 256 + threadIdx.x;
  if (i >= total) return;
  int sp = (int)(i % 512);
  int c  = (int)((i / 512) % 2);
  int n  = (int)(i / 1024);
  float s = bia[c];
  for (int ci = 0; ci < 32; ++ci)
    s += f[((long)n * 32 + ci) * 512 + sp] * w[c * 32 + ci];
  y[i] = s;
}

// ---------------- host side -------------------------------------------------

static inline unsigned gblk(long n) { return (unsigned)((n + 255) / 256); }

static void launch_gemm(hipStream_t st, const bf16_t* A, const bf16_t* B,
                        float* C, bf16_t* Cb, const float* bias,
                        int M, int N, int K, int lda, int ldb, int ldbN, int ldc,
                        long aZb, long aZh, long bZb, long bZh, long cZb, long cZh,
                        int heads, int Z, int flags, int bMode)
{
  dim3 g((unsigned)(N / GTN), (unsigned)(M / GTM), (unsigned)Z);
  k_wmma_gemm<<<g, 256, 0, st>>>(A, B, C, Cb, bias, M, N, K, lda, ldb, ldbN, ldc,
                                 aZb, aZh, bZb, bZh, cZb, cZh, heads, flags, bMode);
}
static void launch_gemm2(hipStream_t st, const bf16_t* A, const bf16_t* B,
                         float* C, bf16_t* Cb, const float* bias,
                         int M, int N, int K, int lda, int ldb, int ldc, int flags)
{
  launch_gemm(st, A, B, C, Cb, bias, M, N, K, lda, ldb, 1, ldc,
              0, 0, 0, 0, 0, 0, 1, 1, flags, 0);
}

// input order = setup_inputs() insertion order, params flattened recursively
enum {
  I_PATCHES = 0, I_COORDS,
  I_C1W, I_C1B, I_BN1G, I_BN1B,
  I_C2W, I_C2B, I_BN2G, I_BN2B,
  I_C3W, I_C3B, I_BN3G, I_BN3B,
  I_PROJW, I_PROJB,
  I_PW1, I_PB1, I_PW2, I_PB2, I_PW3, I_PB3,
  I_WQKV, I_BQKV, I_WO, I_BO, I_LN1G, I_LN1B, I_LN2G, I_LN2B,
  I_W1, I_B1, I_W2, I_B2,
  I_DPW, I_DPB, I_DT1W, I_DT1B, I_DBN1G, I_DBN1B,
  I_DT2W, I_DT2B, I_DBN2G, I_DBN2B,
  I_DT3W, I_DT3B, I_DBN3G, I_DBN3B,
  I_OUTW, I_OUTB,
  I_NPREG, I_NPREB, I_NPOSTG, I_NPOSTB,
  I_BS
};

extern "C" void kernel_launch(void* const* d_in, const int* in_sizes, int n_in,
                              void* d_out, int out_size, void* d_ws, size_t ws_size,
                              hipStream_t stream)
{
  (void)in_sizes; (void)n_in; (void)out_size; (void)ws_size;
  const int BK = 4096, NL = 6, NHh = 8, KS = 1024, Bb = 4;
#define FP(i) ((const float*)d_in[i])

  // ---- workspace layout (arenas reused across phases) ----
  char* ws = (char*)d_ws;
  size_t off = 0;
  auto alloc = [&](size_t bytes) -> char* {
    char* p = ws + off;
    off += (bytes + 255) & ~(size_t)255;
    return p;
  };
  float*  R1      = (float*)alloc(268435456);  // a1 | scores | d3
  float*  R2f     = (float*)alloc(67108864);   // a2 | probs(bf16) | d2
  float*  R3      = (float*)alloc(16777216);   // d1
  float*  tokens  = (float*)alloc(4194304);
  float*  posout  = (float*)alloc(4194304);
  bf16_t* hb      = (bf16_t*)alloc(1048576);
  bf16_t* hb2     = (bf16_t*)alloc(1048576);
  float*  x       = (float*)alloc(4194304);
  bf16_t* xb      = (bf16_t*)alloc(2097152);
  bf16_t* qkvb    = (bf16_t*)alloc(6291456);
  bf16_t* ffhb    = (bf16_t*)alloc(8388608);
  float*  a3      = (float*)alloc(4194304);
  float*  decf    = (float*)alloc(4194304);
  bf16_t* coordsb = (bf16_t*)alloc(262144);    // [BK,32] padded
  float*  stat    = (float*)alloc(4096);
  bf16_t* wt_tok  = (bf16_t*)alloc(131072);
  bf16_t* wt_p1   = (bf16_t*)alloc(8192);      // [32,128] padded
  bf16_t* wt_p2   = (bf16_t*)alloc(32768);
  bf16_t* wt_p3   = (bf16_t*)alloc(65536);
  bf16_t* wt_qkv  = (bf16_t*)alloc(2359296);
  bf16_t* wt_wo   = (bf16_t*)alloc(786432);
  bf16_t* wt_w1   = (bf16_t*)alloc(3145728);
  bf16_t* wt_w2   = (bf16_t*)alloc(3145728);
  bf16_t* wt_dp   = (bf16_t*)alloc(131072);

  // ---- pack weights: W[N,K] f32 -> Wt[K,N] bf16 (transpose, pad) ----
  k_wt<<<dim3(gblk(256L*256), 1), 256, 0, stream>>>(FP(I_PROJW), wt_tok, 256, 256, 256);
  k_wt<<<dim3(gblk(32L*128), 1), 256, 0, stream>>>(FP(I_PW1), wt_p1, 128, 3, 32);
  k_wt<<<dim3(gblk(128L*128), 1), 256, 0, stream>>>(FP(I_PW2), wt_p2, 128, 128, 128);
  k_wt<<<dim3(gblk(128L*256), 1), 256, 0, stream>>>(FP(I_PW3), wt_p3, 256, 128, 128);
  k_wt<<<dim3(gblk(256L*768), NL), 256, 0, stream>>>(FP(I_WQKV), wt_qkv, 768, 256, 256);
  k_wt<<<dim3(gblk(256L*256), NL), 256, 0, stream>>>(FP(I_WO), wt_wo, 256, 256, 256);
  k_wt<<<dim3(gblk(256L*1024), NL), 256, 0, stream>>>(FP(I_W1), wt_w1, 1024, 256, 256);
  k_wt<<<dim3(gblk(1024L*256), NL), 256, 0, stream>>>(FP(I_W2), wt_w2, 256, 1024, 1024);
  k_wt<<<dim3(gblk(256L*256), 1), 256, 0, stream>>>(FP(I_DPW), wt_dp, 256, 256, 256);
  k_pad_coords<<<gblk((long)BK * 32), 256, 0, stream>>>(FP(I_COORDS), coordsb, BK);

  // ---- tokenizer ----
  float* a1 = R1;
  float* a2 = R2f;
  k_conv3<<<gblk((long)BK * 64 * 64), 256, 0, stream>>>(FP(I_PATCHES), FP(I_C1W), FP(I_C1B), a1, BK, 1, 8, 64);
  k_bn_stats<<<64, 256, 0, stream>>>(a1, stat, 64, 64, BK);
  k_bn_gelu<<<gblk((long)BK * 64 * 64), 256, 0, stream>>>(a1, stat, FP(I_BN1G), FP(I_BN1B), (bf16_t*)nullptr, 64, 64, (long)BK * 64 * 64);
  k_conv3<<<gblk((long)BK * 128 * 8), 256, 0, stream>>>(a1, FP(I_C2W), FP(I_C2B), a2, BK, 64, 4, 128);
  k_bn_stats<<<128, 256, 0, stream>>>(a2, stat, 128, 8, BK);
  k_bn_gelu<<<gblk((long)BK * 128 * 8), 256, 0, stream>>>(a2, stat, FP(I_BN2G), FP(I_BN2B), (bf16_t*)nullptr, 128, 8, (long)BK * 128 * 8);
  k_conv_k2<<<gblk((long)BK * 256), 256, 0, stream>>>(a2, FP(I_C3W), FP(I_C3B), a3, BK, 128, 256);
  k_bn_stats<<<256, 256, 0, stream>>>(a3, stat, 256, 1, BK);
  k_bn_gelu<<<gblk((long)BK * 256), 256, 0, stream>>>(a3, stat, FP(I_BN3G), FP(I_BN3B), xb, 256, 1, (long)BK * 256);
  launch_gemm2(stream, xb, wt_tok, tokens, nullptr, FP(I_PROJB), BK, 256, 256, 256, 256, 256, FL_F32);

  // ---- positional MLP (K padded to 32 for the first layer) ----
  launch_gemm2(stream, coordsb, wt_p1, nullptr, hb, FP(I_PB1), BK, 128, 32, 32, 128, 128, FL_GELU | FL_BF16);
  launch_gemm2(stream, hb, wt_p2, nullptr, hb2, FP(I_PB2), BK, 128, 128, 128, 128, 128, FL_GELU | FL_BF16);
  launch_gemm2(stream, hb2, wt_p3, posout, nullptr, FP(I_PB3), BK, 256, 128, 128, 256, 256, FL_F32);

  k_add<<<gblk((long)BK * 256), 256, 0, stream>>>(tokens, posout, x, (long)BK * 256);
  k_layernorm<<<BK, 256, 0, stream>>>(x, FP(I_NPREG), FP(I_NPREB), x, (bf16_t*)nullptr);

  // ---- transformer ----
  float*  scores = R1;
  bf16_t* probs  = (bf16_t*)R2f;
  for (int l = 0; l < NL; ++l) {
    const bf16_t* Wqkv = wt_qkv + (long)l * 256 * 768;
    const bf16_t* Wo   = wt_wo  + (long)l * 256 * 256;
    const bf16_t* W1   = wt_w1  + (long)l * 256 * 1024;
    const bf16_t* W2   = wt_w2  + (long)l * 1024 * 256;
    k_layernorm<<<BK, 256, 0, stream>>>(x, FP(I_LN1G) + l * 256, FP(I_LN1B) + l * 256, (float*)nullptr, xb);
    // qkv (bf16 out direct)
    launch_gemm2(stream, xb, Wqkv, nullptr, qkvb, FP(I_BQKV) + l * 768, BK, 768, 256, 256, 768, 768, FL_BF16);
    // scores[b,h] = Q (1024x32) * K^T (32x1024); K^T read k-contiguous (bMode1)
    launch_gemm(stream, qkvb, qkvb + 256, scores, nullptr, nullptr,
                KS, KS, 32, 768, 0, 768, KS,
                (long)KS * 768, 32, (long)KS * 768, 32,
                (long)NHh * KS * KS, (long)KS * KS, NHh, Bb * NHh, FL_F32, 1);
    k_softmax<<<Bb * NHh * KS, 256, 0, stream>>>(scores, probs, 0.17677669529663687f);
    // attn_out[b,q,h*32+d] = P (1024x1024) * V (1024x32), bf16 out -> xb
    launch_gemm(stream, probs, qkvb + 512, nullptr, xb, nullptr,
                KS, 32, KS, KS, 768, 1, 256,
                (long)NHh * KS * KS, (long)KS * KS, (long)KS * 768, 32,
                (long)KS * 256, 32, NHh, Bb * NHh, FL_BF16, 0);
    // x += attn_out @ Wo^T + bo  (residual fused)
    launch_gemm2(stream, xb, Wo, x, nullptr, FP(I_BO) + l * 256, BK, 256, 256, 256, 256, 256, FL_ACC | FL_F32);
    k_layernorm<<<BK, 256, 0, stream>>>(x, FP(I_LN2G) + l * 256, FP(I_LN2B) + l * 256, (float*)nullptr, xb);
    // h = gelu(x @ W1^T + b1) (bf16 out direct)
    launch_gemm2(stream, xb, W1, nullptr, ffhb, FP(I_B1) + l * 1024, BK, 1024, 256, 256, 1024, 1024, FL_GELU | FL_BF16);
    // x += h @ W2^T + b2
    launch_gemm2(stream, ffhb, W2, x, nullptr, FP(I_B2) + l * 256, BK, 256, 1024, 1024, 256, 256, FL_ACC | FL_F32);
  }
  k_layernorm<<<BK, 256, 0, stream>>>(x, FP(I_NPOSTG), FP(I_NPOSTB), (float*)nullptr, xb);

  // ---- decoder ----
  launch_gemm2(stream, xb, wt_dp, decf, nullptr, FP(I_DPB), BK, 256, 256, 256, 256, 256, FL_F32);
  float* dd1 = R3;
  float* dd2 = R2f;
  float* dd3 = R1;
  k_deconv2x<<<gblk((long)BK * 128 * 8), 256, 0, stream>>>(decf, FP(I_DT1W), FP(I_DT1B), dd1, BK, 256, 1, 128);
  k_bn_stats<<<128, 256, 0, stream>>>(dd1, stat, 128, 8, BK);
  k_bn_gelu<<<gblk((long)BK * 128 * 8), 256, 0, stream>>>(dd1, stat, FP(I_DBN1G), FP(I_DBN1B), (bf16_t*)nullptr, 128, 8, (long)BK * 128 * 8);
  k_deconv2x<<<gblk((long)BK * 64 * 64), 256, 0, stream>>>(dd1, FP(I_DT2W), FP(I_DT2B), dd2, BK, 128, 2, 64);
  k_bn_stats<<<64, 256, 0, stream>>>(dd2, stat, 64, 64, BK);
  k_bn_gelu<<<gblk((long)BK * 64 * 64), 256, 0, stream>>>(dd2, stat, FP(I_DBN2G), FP(I_DBN2B), (bf16_t*)nullptr, 64, 64, (long)BK * 64 * 64);
  k_deconv2x<<<gblk((long)BK * 32 * 512), 256, 0, stream>>>(dd2, FP(I_DT3W), FP(I_DT3B), dd3, BK, 64, 4, 32);
  k_bn_stats<<<32, 256, 0, stream>>>(dd3, stat, 32, 512, BK);
  k_bn_gelu<<<gblk((long)BK * 32 * 512), 256, 0, stream>>>(dd3, stat, FP(I_DBN3G), FP(I_DBN3B), (bf16_t*)nullptr, 32, 512, (long)BK * 32 * 512);
  k_outconv<<<gblk((long)BK * 2 * 512), 256, 0, stream>>>(dd3, FP(I_OUTW), FP(I_OUTB), (float*)d_out, BK);
#undef FP
}
